// SwitchTransformerLayer_48576080118268
// MI455X (gfx1250) — matrix operand order repaired and checked by reading, not compiled, not run on hardware
//
#include <hip/hip_runtime.h>
#include <hip/hip_bf16.h>
#include <math.h>

typedef __bf16 bf16_t;
typedef __attribute__((ext_vector_type(16))) __bf16 v16bf;
typedef __attribute__((ext_vector_type(8)))  __bf16 v8bf;
typedef __attribute__((ext_vector_type(4)))  __bf16 v4bf;
typedef __attribute__((ext_vector_type(8)))  float   v8f;
typedef __attribute__((ext_vector_type(4)))  float   v4f;

#define NEXP 8
#define DIM  1024
#define FF   4096
#define NTOK 8192
#define CAP  1536   // ceil(1.5 * 8192 / 8)

// LDS tile row stride in bf16 elements: 40 -> 80 bytes (16B aligned, spreads banks)
#define TSTR 40

static __device__ __forceinline__ v16bf cat16(v8bf lo, v8bf hi) {
  return __builtin_shufflevector(lo, hi, 0,1,2,3,4,5,6,7,8,9,10,11,12,13,14,15);
}

static __device__ __forceinline__ v8f vzero8() {
  v8f z = {0.f,0.f,0.f,0.f,0.f,0.f,0.f,0.f};
  return z;
}

// low 32 bits of a generic shared pointer == LDS byte address
static __device__ __forceinline__ unsigned lds_addr_of(const void* p) {
  return (unsigned)(unsigned long long)p;
}

static __device__ __forceinline__ void async_cp16(unsigned lds_dst, const void* gsrc) {
  unsigned long long ga = (unsigned long long)gsrc;
  asm volatile("global_load_async_to_lds_b128 %0, %1, off"
               :: "v"(lds_dst), "v"(ga) : "memory");
}
static __device__ __forceinline__ void async_wait0() {
  asm volatile("s_wait_asynccnt 0x0" ::: "memory");
}

// ---------------------------------------------------------------------------
// 0) zero output, invalidate slot->token map
// ---------------------------------------------------------------------------
__global__ void init_kernel(float* __restrict__ y, int* __restrict__ tok_of) {
  size_t i0 = (size_t)blockIdx.x * blockDim.x + threadIdx.x;
  size_t stride = (size_t)gridDim.x * blockDim.x;
  const size_t ND = (size_t)NTOK * DIM;
  for (size_t i = i0; i < ND; i += stride) y[i] = 0.0f;
  const size_t EC = (size_t)NEXP * CAP;
  for (size_t i = i0; i < EC; i += stride) tok_of[i] = -1;
}

// ---------------------------------------------------------------------------
// 1) router: logits -> top-1 expert + its softmax prob (= 1/sum(exp(l-lmax)))
// ---------------------------------------------------------------------------
__global__ __launch_bounds__(256)
void router_kernel(const float* __restrict__ x, const float* __restrict__ rw,
                   int* __restrict__ chosen_e, float* __restrict__ prob) {
  __shared__ float red[256 * NEXP];
  const int t   = blockIdx.x;
  const int tid = threadIdx.x;
  float acc[NEXP];
#pragma unroll
  for (int e = 0; e < NEXP; ++e) acc[e] = 0.f;
  const float* xr = x + (size_t)t * DIM;
  for (int d = tid; d < DIM; d += 256) {
    float xv = xr[d];
#pragma unroll
    for (int e = 0; e < NEXP; ++e) acc[e] += xv * rw[e * DIM + d];
  }
#pragma unroll
  for (int e = 0; e < NEXP; ++e) red[tid * NEXP + e] = acc[e];
  __syncthreads();
  for (int s = 128; s > 0; s >>= 1) {
    if (tid < s) {
#pragma unroll
      for (int e = 0; e < NEXP; ++e) red[tid * NEXP + e] += red[(tid + s) * NEXP + e];
    }
    __syncthreads();
  }
  if (tid == 0) {
    float best = red[0]; int be = 0;
#pragma unroll
    for (int e = 1; e < NEXP; ++e) { float v = red[e]; if (v > best) { best = v; be = e; } }
    float sum = 0.f;
#pragma unroll
    for (int e = 0; e < NEXP; ++e) sum += expf(red[e] - best);
    chosen_e[t] = be;
    prob[t] = 1.0f / sum;   // max softmax prob
  }
}

// ---------------------------------------------------------------------------
// 2) FCFS capacity assignment (one wave32, ballot prefix within 32-token chunks)
// ---------------------------------------------------------------------------
__global__ void assign_kernel(const int* __restrict__ chosen_e, int* __restrict__ tok_of) {
  const int lane = threadIdx.x;
  int cnt[NEXP];
#pragma unroll
  for (int e = 0; e < NEXP; ++e) cnt[e] = 0;
  const unsigned lt = (lane == 0) ? 0u : ((1u << lane) - 1u);
  for (int base = 0; base < NTOK; base += 32) {
    int ce = chosen_e[base + lane];
#pragma unroll
    for (int e = 0; e < NEXP; ++e) {
      unsigned mask = (unsigned)__ballot(ce == e);
      if (ce == e) {
        int p = cnt[e] + __popc(mask & lt);
        if (p < CAP) tok_of[e * CAP + p] = base + lane;
      }
      cnt[e] += __popc(mask);
    }
  }
}

// ---------------------------------------------------------------------------
// 2b) weight prep: transpose + convert  src[e][R][C] f32 -> dst[e][C][R] bf16
//     (one streaming pass; removes all cvt + scalar LDS stores from GEMMs)
// ---------------------------------------------------------------------------
__global__ __launch_bounds__(256)
void transpose_cvt_kernel(const float* __restrict__ src, bf16_t* __restrict__ dst,
                          int R, int C) {
  __shared__ bf16_t T[32][33];
  const int e  = blockIdx.z;
  const int r0 = blockIdx.x * 32;
  const int c0 = blockIdx.y * 32;
  const int tid = threadIdx.x;
  const int r = tid >> 3;        // 0..31
  const int q = tid & 7;         // 0..7  (4 columns each)

  const float* s = src + (size_t)e * R * C;
  bf16_t*      d = dst + (size_t)e * R * C;

  v4f v = *(const v4f*)(s + (size_t)(r0 + r) * C + c0 + q * 4);
#pragma unroll
  for (int j = 0; j < 4; ++j) T[q * 4 + j][r] = (bf16_t)v[j];
  __syncthreads();
  // dst[(c0+cc)*R + r0 + rr] = T[cc][rr]; thread writes row cc=r, cols rr=q*4..+3
  bf16_t o0 = T[r][q * 4 + 0];
  bf16_t o1 = T[r][q * 4 + 1];
  bf16_t o2 = T[r][q * 4 + 2];
  bf16_t o3 = T[r][q * 4 + 3];
  v4bf o; o[0] = o0; o[1] = o1; o[2] = o2; o[3] = o3;
  *(v4bf*)(d + (size_t)(c0 + r) * R + r0 + q * 4) = o;
}

// ---------------------------------------------------------------------------
// 3) expert GEMM1: h = gelu(gather(x) @ w1 + b1), bf16 out.
//    A: gathered x rows (f32->bf16 in regs). B: async bf16 copy from w1t [F][D].
// ---------------------------------------------------------------------------
__global__ __launch_bounds__(256)
void moe_ffn1_kernel(const float* __restrict__ x, const bf16_t* __restrict__ w1t,
                     const float* __restrict__ b1, const int* __restrict__ tok_of,
                     bf16_t* __restrict__ h) {
  const int e    = blockIdx.z;
  const int c0   = blockIdx.y * 128;
  const int f0   = blockIdx.x * 128;
  const int tid  = threadIdx.x;
  const int lane = tid & 31;
  const int wave = tid >> 5;
  const int wm   = wave >> 1;
  const int wn   = wave & 1;

  __shared__ bf16_t At[2][128 * TSTR];
  __shared__ bf16_t Bt[2][128 * TSTR];
  __shared__ int    stok[128];

  if (tid < 128) stok[tid] = tok_of[e * CAP + c0 + tid];
  __syncthreads();

  // A: 4 float4 chunks/thread: row = lin>>3, kc = lin&7
  const float* asrc[4];
  int arow[4], akc[4];
#pragma unroll
  for (int i = 0; i < 4; ++i) {
    int lin = tid + 256 * i;
    arow[i] = lin >> 3;
    akc[i]  = lin & 7;
    int tok = stok[arow[i]];
    asrc[i] = (tok >= 0) ? (x + (size_t)tok * DIM + akc[i] * 4) : (const float*)0;
  }
  // B: 2 16-byte bf16 chunks/thread from w1t[f][k]: n = lin>>2, kc = lin&3
  const bf16_t* btsrc[2];
  int bn[2], bkc[2];
  const bf16_t* w1te = w1t + (size_t)e * FF * DIM;
#pragma unroll
  for (int i = 0; i < 2; ++i) {
    int lin = tid + 256 * i;
    bn[i]  = lin >> 2;
    bkc[i] = lin & 3;
    btsrc[i] = w1te + (size_t)(f0 + bn[i]) * DIM + bkc[i] * 8;
  }

  v8f acc[2][4];
#pragma unroll
  for (int i = 0; i < 2; ++i)
#pragma unroll
    for (int j = 0; j < 4; ++j) acc[i][j] = vzero8();

  const int hi = lane >> 4;
  const int lr = lane & 15;
  const int NK = DIM / 32;

  v4f ar[4];
  // ---- prologue: tile 0 -> buffer 0 ----
#pragma unroll
  for (int i = 0; i < 2; ++i)
    async_cp16(lds_addr_of(&Bt[0][bn[i] * TSTR + bkc[i] * 8]), btsrc[i]);
#pragma unroll
  for (int i = 0; i < 4; ++i) {
    v4f v = {0.f, 0.f, 0.f, 0.f};
    if (asrc[i]) v = *(const v4f*)(asrc[i]);
    ar[i] = v;
  }
#pragma unroll
  for (int i = 0; i < 4; ++i) {
    v4bf b;
    b[0]=(bf16_t)ar[i][0]; b[1]=(bf16_t)ar[i][1]; b[2]=(bf16_t)ar[i][2]; b[3]=(bf16_t)ar[i][3];
    *(v4bf*)(&At[0][arow[i] * TSTR + akc[i] * 4]) = b;
  }
  async_wait0();
  __syncthreads();

  for (int kt = 0; kt < NK; ++kt) {
    const int cur = kt & 1;
    const int nxt = cur ^ 1;
    const bool more = (kt + 1) < NK;
    const int knext = (kt + 1) * 32;

    if (more) {
#pragma unroll
      for (int i = 0; i < 2; ++i)
        async_cp16(lds_addr_of(&Bt[nxt][bn[i] * TSTR + bkc[i] * 8]), btsrc[i] + knext);
#pragma unroll
      for (int i = 0; i < 4; ++i) {
        v4f v = {0.f, 0.f, 0.f, 0.f};
        if (asrc[i]) {
          v = *(const v4f*)(asrc[i] + knext);
          if (kt + 2 < NK) __builtin_prefetch(asrc[i] + knext + 32, 0, 1);
        }
        ar[i] = v;
      }
    }

    // ---- compute on current buffer ----
    v16bf af[2];
#pragma unroll
    for (int mt = 0; mt < 2; ++mt) {
      const bf16_t* arr = &At[cur][(wm * 32 + mt * 16 + lr) * TSTR];
      v8bf lo = *(const v8bf*)(arr + 8 * hi);
      v8bf hh = *(const v8bf*)(arr + 16 + 8 * hi);
      af[mt] = cat16(lo, hh);
    }
    v16bf bfm[4];
#pragma unroll
    for (int nt = 0; nt < 4; ++nt) {
      const bf16_t* brr = &Bt[cur][(wn * 64 + nt * 16 + lr) * TSTR];
      v8bf lo = *(const v8bf*)(brr + 8 * hi);
      v8bf hh = *(const v8bf*)(brr + 16 + 8 * hi);
      bfm[nt] = cat16(lo, hh);
    }
#pragma unroll
    for (int mt = 0; mt < 2; ++mt)
#pragma unroll
      for (int nt = 0; nt < 4; ++nt)
        acc[mt][nt] = __builtin_amdgcn_wmma_f32_16x16x32_bf16(
            false, af[mt], false, bfm[nt], (short)0, acc[mt][nt], false, false);

    if (more) {
#pragma unroll
      for (int i = 0; i < 4; ++i) {
        v4bf b;
        b[0]=(bf16_t)ar[i][0]; b[1]=(bf16_t)ar[i][1]; b[2]=(bf16_t)ar[i][2]; b[3]=(bf16_t)ar[i][3];
        *(v4bf*)(&At[nxt][arow[i] * TSTR + akc[i] * 4]) = b;
      }
      async_wait0();
    }
    __syncthreads();
  }

  // ---- epilogue: bias + exact GELU, store bf16 h ----
  const int n = lane & 15;
#pragma unroll
  for (int nt = 0; nt < 4; ++nt) {
    int f = f0 + wn * 64 + nt * 16 + n;
    float bb = b1[e * FF + f];
#pragma unroll
    for (int mt = 0; mt < 2; ++mt) {
#pragma unroll
      for (int r = 0; r < 8; ++r) {
        int c = c0 + wm * 32 + mt * 16 + 8 * hi + r;
        float v = acc[mt][nt][r] + bb;
        v = 0.5f * v * (1.0f + erff(v * 0.70710678118654752f));
        h[((size_t)e * CAP + c) * FF + f] = (bf16_t)v;
      }
    }
  }
}

// ---------------------------------------------------------------------------
// 4) expert GEMM2 + fused scatter. Fully async data movement: A tile from h,
//    B tile from w2t, both bf16 16B chunks on ASYNCcnt; hot loop = ds+wmma.
// ---------------------------------------------------------------------------
__global__ __launch_bounds__(256)
void moe_ffn2_kernel(const bf16_t* __restrict__ h, const bf16_t* __restrict__ w2t,
                     const float* __restrict__ b2, const int* __restrict__ tok_of,
                     const float* __restrict__ prob, float* __restrict__ y) {
  const int e    = blockIdx.z;
  const int c0   = blockIdx.y * 128;
  const int n0   = blockIdx.x * 128;
  const int tid  = threadIdx.x;
  const int lane = tid & 31;
  const int wave = tid >> 5;
  const int wm   = wave >> 1;
  const int wn   = wave & 1;

  __shared__ bf16_t At[2][128 * TSTR];
  __shared__ bf16_t Bt[2][128 * TSTR];
  __shared__ int    stok[128];
  __shared__ float  sprob[128];

  if (tid < 128) {
    int t = tok_of[e * CAP + c0 + tid];
    stok[tid]  = t;
    sprob[tid] = (t >= 0) ? prob[t] : 0.f;
  }
  __syncthreads();

  // A: 2 chunks/thread of 8 bf16 (16B): row = lin>>2, kc = lin&3
  const bf16_t* hsrc[2];
  int arow[2], akc[2];
#pragma unroll
  for (int i = 0; i < 2; ++i) {
    int lin = tid + 256 * i;
    arow[i] = lin >> 2;
    akc[i]  = lin & 3;
    hsrc[i] = h + ((size_t)e * CAP + c0 + arow[i]) * FF + akc[i] * 8;
  }
  // B: 2 chunks/thread from w2t[n][k] (K = FF): n = lin>>2, kc = lin&3
  const bf16_t* btsrc[2];
  int bn[2], bkc[2];
  const bf16_t* w2te = w2t + (size_t)e * FF * DIM;
#pragma unroll
  for (int i = 0; i < 2; ++i) {
    int lin = tid + 256 * i;
    bn[i]  = lin >> 2;
    bkc[i] = lin & 3;
    btsrc[i] = w2te + (size_t)(n0 + bn[i]) * FF + bkc[i] * 8;
  }

  v8f acc[2][4];
#pragma unroll
  for (int i = 0; i < 2; ++i)
#pragma unroll
    for (int j = 0; j < 4; ++j) acc[i][j] = vzero8();

  const int hi = lane >> 4;
  const int lr = lane & 15;
  const int NK = FF / 32;

  // ---- prologue: tile 0 -> buffer 0 (all async) ----
#pragma unroll
  for (int i = 0; i < 2; ++i) {
    async_cp16(lds_addr_of(&At[0][arow[i] * TSTR + akc[i] * 8]), hsrc[i]);
    async_cp16(lds_addr_of(&Bt[0][bn[i] * TSTR + bkc[i] * 8]), btsrc[i]);
  }
  async_wait0();
  __syncthreads();

  for (int kt = 0; kt < NK; ++kt) {
    const int cur = kt & 1;
    const int nxt = cur ^ 1;
    const bool more = (kt + 1) < NK;
    const int knext = (kt + 1) * 32;

    if (more) {
#pragma unroll
      for (int i = 0; i < 2; ++i) {
        async_cp16(lds_addr_of(&At[nxt][arow[i] * TSTR + akc[i] * 8]), hsrc[i] + knext);
        async_cp16(lds_addr_of(&Bt[nxt][bn[i] * TSTR + bkc[i] * 8]), btsrc[i] + knext);
      }
    }

    // ---- compute on current buffer ----
    v16bf af[2];
#pragma unroll
    for (int mt = 0; mt < 2; ++mt) {
      const bf16_t* arr = &At[cur][(wm * 32 + mt * 16 + lr) * TSTR];
      v8bf lo = *(const v8bf*)(arr + 8 * hi);
      v8bf hh = *(const v8bf*)(arr + 16 + 8 * hi);
      af[mt] = cat16(lo, hh);
    }
    v16bf bfm[4];
#pragma unroll
    for (int nt = 0; nt < 4; ++nt) {
      const bf16_t* brr = &Bt[cur][(wn * 64 + nt * 16 + lr) * TSTR];
      v8bf lo = *(const v8bf*)(brr + 8 * hi);
      v8bf hh = *(const v8bf*)(brr + 16 + 8 * hi);
      bfm[nt] = cat16(lo, hh);
    }
#pragma unroll
    for (int mt = 0; mt < 2; ++mt)
#pragma unroll
      for (int nt = 0; nt < 4; ++nt)
        acc[mt][nt] = __builtin_amdgcn_wmma_f32_16x16x32_bf16(
            false, af[mt], false, bfm[nt], (short)0, acc[mt][nt], false, false);

    if (more) async_wait0();
    __syncthreads();
  }

  // ---- epilogue: bias, scale by router prob, scatter to y[tok] ----
  const int n = lane & 15;
#pragma unroll
  for (int nt = 0; nt < 4; ++nt) {
    int col = n0 + wn * 64 + nt * 16 + n;
    float bb = b2[e * DIM + col];
#pragma unroll
    for (int mt = 0; mt < 2; ++mt) {
#pragma unroll
      for (int r = 0; r < 8; ++r) {
        int cl = wm * 32 + mt * 16 + 8 * hi + r;
        int tok = stok[cl];
        if (tok >= 0) {
          y[(size_t)tok * DIM + col] = (acc[mt][nt][r] + bb) * sprob[cl];
        }
      }
    }
  }
}

// ---------------------------------------------------------------------------
extern "C" void kernel_launch(void* const* d_in, const int* in_sizes, int n_in,
                              void* d_out, int out_size, void* d_ws, size_t ws_size,
                              hipStream_t stream) {
  const float* x  = (const float*)d_in[0];
  const float* rw = (const float*)d_in[1];
  const float* w1 = (const float*)d_in[2];
  const float* b1 = (const float*)d_in[3];
  const float* w2 = (const float*)d_in[4];
  const float* b2 = (const float*)d_in[5];
  float* y = (float*)d_out;

  char* ws = (char*)d_ws;
  int* chosen  = (int*)ws;     ws += (((size_t)NTOK * 4) + 255) & ~(size_t)255;
  float* prob  = (float*)ws;   ws += (((size_t)NTOK * 4) + 255) & ~(size_t)255;
  int* tok_of  = (int*)ws;     ws += (((size_t)NEXP * CAP * 4) + 255) & ~(size_t)255;
  bf16_t* h    = (bf16_t*)ws;  ws += (size_t)NEXP * CAP * FF * 2;   // ~96 MB
  bf16_t* w1t  = (bf16_t*)ws;  ws += (size_t)NEXP * DIM * FF * 2;   // 64 MB
  bf16_t* w2t  = (bf16_t*)ws;                                       // 64 MB

  init_kernel<<<2048, 256, 0, stream>>>(y, tok_of);
  router_kernel<<<NTOK, 256, 0, stream>>>(x, rw, chosen, prob);
  assign_kernel<<<1, 32, 0, stream>>>(chosen, tok_of);
  // w1[e][D][F] -> w1t[e][F][D];  w2[e][F][D] -> w2t[e][D][F]
  transpose_cvt_kernel<<<dim3(DIM / 32, FF / 32, NEXP), 256, 0, stream>>>(w1, w1t, DIM, FF);
  transpose_cvt_kernel<<<dim3(FF / 32, DIM / 32, NEXP), 256, 0, stream>>>(w2, w2t, FF, DIM);
  moe_ffn1_kernel<<<dim3(FF / 128, CAP / 128, NEXP), 256, 0, stream>>>(x, w1t, b1, tok_of, h);
  moe_ffn2_kernel<<<dim3(DIM / 128, CAP / 128, NEXP), 256, 0, stream>>>(h, w2t, b2, tok_of, prob, y);
}
